// LSTMDecoder_63763084477032
// MI455X (gfx1250) — matrix operand order repaired and checked by reading, not compiled, not run on hardware
//
#include <hip/hip_runtime.h>

// ---------------------------------------------------------------------------
// LSTM decoder, fused per-cell WMMA kernel for gfx1250 (MI455X).
//   L=5, H=1024, DOUT=1, T=128, BS=512
// v2: async global->LDS weight staging (ASYNCcnt) + double-buffered LDS +
//     software-pipelined A-fragment loads. One barrier per K-chunk.
// ---------------------------------------------------------------------------

#define LSTM_L   5
#define LSTM_H   1024
#define LSTM_BS  512
#define LSTM_T   128
#define ROWK     40          // padded K-row stride (elements) in LDS B tile

typedef __attribute__((ext_vector_type(16))) __bf16 v16bf;
typedef __attribute__((ext_vector_type(8)))  __bf16 v8bf;
typedef __attribute__((ext_vector_type(8)))  float  v8f;

__device__ __forceinline__ float sigm_f(float x) {
    return 1.0f / (1.0f + __expf(-x));
}
__device__ __forceinline__ float tanh_f(float x) {
    float ax = fabsf(x);
    float e  = __expf(2.0f * ax);
    float t  = (e - 1.0f) / (e + 1.0f);
    t = (ax > 15.0f) ? 1.0f : t;
    return copysignf(t, x);
}

__device__ __forceinline__ v16bf cat8(v8bf lo, v8bf hi) {
    return __builtin_shufflevector(lo, hi,
        0, 1, 2, 3, 4, 5, 6, 7, 8, 9, 10, 11, 12, 13, 14, 15);
}

// 32-bit LDS byte offset of a generic pointer into shared memory
__device__ __forceinline__ unsigned lds_offset(const void* p) {
    return (unsigned)(unsigned long long)
           (__attribute__((address_space(3))) const void*)p;
}

// CDNA5 async global -> LDS copy, 16 bytes per lane (tracked with ASYNCcnt)
__device__ __forceinline__ void async_copy_b128(unsigned ldsOff,
                                                const void* gaddr) {
    asm volatile("global_load_async_to_lds_b128 %0, %1, off"
                 :: "v"(ldsOff), "v"((unsigned long long)gaddr)
                 : "memory");
}
__device__ __forceinline__ void wait_asynccnt0() {
    asm volatile("s_wait_asynccnt 0" ::: "memory");
}
__device__ __forceinline__ void wait_dscnt0() {
    asm volatile("s_wait_dscnt 0" ::: "memory");
}

struct AFrag { v8bf lo, hi; };

// ---------------------------------------------------------------------------
// Fused LSTM cell: one launch per (t, layer).
//   grid = (4, 64): blockIdx.x -> 128-row batch block, blockIdx.y -> 16 h cols
//   block = 256 threads = 8 waves; wave w handles batch rows mBase..mBase+15.
// Each wave keeps the SAME 16 hidden columns across all four gate regions
// (n = g*1024 + h) so i/f/g/o live in its 4 accumulators and the nonlinearity
// plus c/h update are done fully in registers.
// Wih/xin == nullptr for layer 0 (scalar input handled in epilogue via w0/inp).
// ---------------------------------------------------------------------------
__global__ void __launch_bounds__(256)
lstm_cell_kernel(const __bf16* __restrict__ Whh,   // (4H, H) bf16
                 const __bf16* __restrict__ Wih,   // (4H, H) bf16 or nullptr
                 const float*  __restrict__ bias,  // (4H) combined b_ih+b_hh
                 const float*  __restrict__ w0,    // (4H) W_ih0[:,0] fp32
                 const float*  __restrict__ inp,   // (BS) scalar input
                 const __bf16* __restrict__ hRead, // (BS,H) own-layer h (t-1)
                 const __bf16* __restrict__ xin,   // (BS,H) layer l-1 h (t)
                 __bf16*       __restrict__ hWrite,// (BS,H) h out (t)
                 float*        __restrict__ cState)// (BS,H) c in/out
{
    __shared__ __bf16 ldsB[2][4 * 16 * ROWK];      // double-buffered B tiles

    const int tid   = threadIdx.x;
    const int wave  = tid >> 5;
    const int lane  = tid & 31;
    const int r     = lane & 15;                   // row/col within tile
    const int half  = lane >> 4;                   // lane-half select
    const int hBase = blockIdx.y * 16;
    const int mBase = blockIdx.x * 128 + wave * 16;

    // staging decomposition: 4 gates x 16 cols x 4 k-quads = 256 threads
    const int sg  = tid >> 6;
    const int sc  = (tid >> 2) & 15;
    const int skq = tid & 3;
    const int sN  = sg * 1024 + hBase + sc;        // gate column in [0,4096)

    const int KC    = LSTM_H / 32;                 // 32 K-chunks per matrix
    const int total = (Wih == nullptr) ? KC : 2 * KC;

    // per-thread LDS destination offsets (bytes) for the two buffers
    const unsigned ldsDst0 = lds_offset(&ldsB[0][(sg * 16 + sc) * ROWK + skq * 8]);
    const unsigned ldsDst1 = lds_offset(&ldsB[1][(sg * 16 + sc) * ROWK + skq * 8]);

    auto b_gaddr = [&](int c) -> const __bf16* {
        const __bf16* Bs = (c < KC) ? Whh : Wih;
        return Bs + (size_t)sN * LSTM_H + ((c & (KC - 1)) << 5) + skq * 8;
    };
    auto a_row = [&](int c) -> const __bf16* {
        const __bf16* As = (c < KC) ? hRead : xin;
        return As + (size_t)(mBase + r) * LSTM_H + ((c & (KC - 1)) << 5);
    };
    auto load_a = [&](int c) -> AFrag {
        const __bf16* p = a_row(c);
        AFrag a;
        a.lo = *(const v8bf*)(p + half * 8);
        a.hi = *(const v8bf*)(p + 16 + half * 8);
        return a;
    };

    v8f acc[4] = {};                               // i, f, g, o accumulators

    // ---- prologue: stage chunk 0 asynchronously, issue A loads for chunk 0
    async_copy_b128(ldsDst0, b_gaddr(0));
    AFrag aCur = load_a(0);

    for (int c = 0; c < total; ++c) {
        const int par = c & 1;

        wait_asynccnt0();                          // own async piece landed
        __syncthreads();                           // whole B chunk visible

        AFrag aNext = aCur;
        if (c + 1 < total) {
            // stream next chunk into the other LDS buffer + next A fragment
            async_copy_b128(par ? ldsDst0 : ldsDst1, b_gaddr(c + 1));
            aNext = load_a(c + 1);
        }

        const v16bf afrag = cat8(aCur.lo, aCur.hi);
        const __bf16* buf = ldsB[par];
#pragma unroll
        for (int g = 0; g < 4; ++g) {
            const int boff = (g * 16 + r) * ROWK + half * 8;
            v8bf blo = *(const v8bf*)&buf[boff];
            v8bf bhi = *(const v8bf*)&buf[boff + 16];
            v16bf bfrag = cat8(blo, bhi);
            acc[g] = __builtin_amdgcn_wmma_f32_16x16x32_bf16(
                false, afrag, false, bfrag, (short)0, acc[g],
                false, false);
        }
        wait_dscnt0();   // all reads of this buffer done before next overwrite
        aCur = aNext;
    }

    // ------------------- epilogue: gate math, in-register -------------------
    const int hcol = hBase + r;
    const float bI = bias[hcol];
    const float bF = bias[1024 + hcol];
    const float bG = bias[2048 + hcol];
    const float bO = bias[3072 + hcol];

    const bool scalarIn = (Wih == nullptr);
    float wI = 0.f, wF = 0.f, wG = 0.f, wO = 0.f;
    if (scalarIn) {
        wI = w0[hcol]; wF = w0[1024 + hcol];
        wG = w0[2048 + hcol]; wO = w0[3072 + hcol];
    }

#pragma unroll
    for (int j = 0; j < 8; ++j) {
        const int m = mBase + half * 8 + j;
        const size_t idx = (size_t)m * LSTM_H + hcol;
        const float xm = scalarIn ? inp[m] : 0.0f;

        const float gi = sigm_f(acc[0][j] + bI + xm * wI);
        const float gf = sigm_f(acc[1][j] + bF + xm * wF);
        const float gg = tanh_f(acc[2][j] + bG + xm * wG);
        const float go = sigm_f(acc[3][j] + bO + xm * wO);

        const float cp = cState[idx];
        const float cn = gf * cp + gi * gg;
        const float hn = go * tanh_f(cn);

        cState[idx] = cn;
        hWrite[idx] = (__bf16)hn;
    }
}

// ---------------------------------------------------------------------------
// Readout: out[b] = h4[b,:] . Wro + bro ; feeds back as next scalar input.
// One wave per batch element. grid = 64 blocks x 256 threads.
// ---------------------------------------------------------------------------
__global__ void __launch_bounds__(256)
lstm_readout_kernel(const __bf16* __restrict__ h4,
                    const float* __restrict__ Wro,
                    const float* __restrict__ bro,
                    float* __restrict__ out,
                    float* __restrict__ inp,
                    int t)
{
    const int b    = blockIdx.x * 8 + (threadIdx.x >> 5);
    const int lane = threadIdx.x & 31;
    float p = 0.0f;
#pragma unroll 4
    for (int i = 0; i < LSTM_H / 32; ++i) {
        const int h = i * 32 + lane;
        p += (float)h4[(size_t)b * LSTM_H + h] * Wro[h];
    }
    for (int off = 16; off > 0; off >>= 1)
        p += __shfl_xor(p, off, 32);
    if (lane == 0) {
        const float o = p + bro[0];
        out[(size_t)b * LSTM_T + t] = o;
        inp[b] = o;
    }
}

// ---------------------------------------------------------------------------
// Init kernels (rerun every call; deterministic).
// ---------------------------------------------------------------------------
__global__ void cvt_bf16_kernel(const float* __restrict__ s,
                                __bf16* __restrict__ d, long n)
{
    long i = (long)blockIdx.x * blockDim.x + threadIdx.x;
    long stride = (long)gridDim.x * blockDim.x;
    for (; i < n; i += stride) d[i] = (__bf16)s[i];
}

__global__ void combine_bias_kernel(const float* __restrict__ bi,
                                    const float* __restrict__ bh,
                                    float* __restrict__ bo, int n)
{
    int i = blockIdx.x * blockDim.x + threadIdx.x;
    if (i < n) bo[i] = bi[i] + bh[i];
}

__global__ void init_state_kernel(const float* __restrict__ x,
                                  __bf16* __restrict__ hA,
                                  float* __restrict__ c,
                                  float* __restrict__ inp)
{
    const long n = (long)LSTM_L * LSTM_BS * LSTM_H;
    long i = (long)blockIdx.x * blockDim.x + threadIdx.x;
    long stride = (long)gridDim.x * blockDim.x;
    for (; i < n; i += stride) {
        const long src = i & ((long)LSTM_BS * LSTM_H - 1);
        const float v = x[src];
        hA[i] = (__bf16)v;
        c[i]  = v;
        if (i < LSTM_BS) inp[i] = 0.0f;
    }
}

// ---------------------------------------------------------------------------
extern "C" void kernel_launch(void* const* d_in, const int* in_sizes, int n_in,
                              void* d_out, int out_size, void* d_ws, size_t ws_size,
                              hipStream_t stream) {
    (void)in_sizes; (void)n_in; (void)out_size; (void)ws_size;

    const float* x        = (const float*)d_in[0];  // (BS,H)
    const float* W_ih0    = (const float*)d_in[1];  // (4H,1)
    const float* W_ih_r   = (const float*)d_in[2];  // (L-1,4H,H)
    const float* W_hh     = (const float*)d_in[3];  // (L,4H,H)
    const float* b_ih     = (const float*)d_in[4];  // (L,4H)
    const float* b_hh     = (const float*)d_in[5];  // (L,4H)
    const float* W_ro     = (const float*)d_in[6];  // (1,H)
    const float* b_ro     = (const float*)d_in[7];  // (1,)
    float* out = (float*)d_out;                     // (BS,T,1)

    const size_t NW = (size_t)4 * LSTM_H * LSTM_H;  // 4H*H elems per layer
    const size_t SH = (size_t)LSTM_BS * LSTM_H;     // state elems per layer

    // workspace carve-out (256B aligned)
    char* ws = (char*)d_ws;
    size_t off = 0;
    auto carve = [&](size_t bytes) -> void* {
        void* p = ws + off;
        off += (bytes + 255) & ~(size_t)255;
        return p;
    };
    __bf16* WhhBf = (__bf16*)carve(LSTM_L * NW * sizeof(__bf16));
    __bf16* WihBf = (__bf16*)carve((LSTM_L - 1) * NW * sizeof(__bf16));
    float*  biasC = (float*)carve(LSTM_L * 4 * LSTM_H * sizeof(float));
    __bf16* hA    = (__bf16*)carve(LSTM_L * SH * sizeof(__bf16));
    __bf16* hB    = (__bf16*)carve(LSTM_L * SH * sizeof(__bf16));
    float*  cSt   = (float*)carve(LSTM_L * SH * sizeof(float));
    float*  inp   = (float*)carve(LSTM_BS * sizeof(float));

    // ---- one-time (per call) prep: bf16 weights, combined bias, state ----
    cvt_bf16_kernel<<<2048, 256, 0, stream>>>(W_hh,   WhhBf, (long)(LSTM_L * NW));
    cvt_bf16_kernel<<<2048, 256, 0, stream>>>(W_ih_r, WihBf, (long)((LSTM_L - 1) * NW));
    combine_bias_kernel<<<(LSTM_L * 4 * LSTM_H + 255) / 256, 256, 0, stream>>>(
        b_ih, b_hh, biasC, LSTM_L * 4 * LSTM_H);
    init_state_kernel<<<2048, 256, 0, stream>>>(x, hA, cSt, inp);

    // ---- sequential T x L cell chain ----
    const dim3 cellGrid(4, 64, 1);
    for (int t = 0; t < LSTM_T; ++t) {
        __bf16* hR = (t & 1) ? hB : hA;
        __bf16* hW = (t & 1) ? hA : hB;
        for (int l = 0; l < LSTM_L; ++l) {
            const __bf16* wih = (l == 0) ? nullptr : WihBf + (size_t)(l - 1) * NW;
            const __bf16* xin = (l == 0) ? nullptr : hW + (size_t)(l - 1) * SH;
            lstm_cell_kernel<<<cellGrid, 256, 0, stream>>>(
                WhhBf + (size_t)l * NW, wih,
                biasC + (size_t)l * 4 * LSTM_H,
                W_ih0, inp,
                hR + (size_t)l * SH, xin,
                hW + (size_t)l * SH,
                cSt + (size_t)l * SH);
        }
        lstm_readout_kernel<<<64, 256, 0, stream>>>(
            hW + (size_t)(LSTM_L - 1) * SH, W_ro, b_ro, out, inp, t);
    }
}
